// MultiHeadAttention_9431748182382
// MI455X (gfx1250) — matrix operand order repaired and checked by reading, not compiled, not run on hardware
//
#include <hip/hip_runtime.h>

typedef _Float16 f16;
typedef __attribute__((ext_vector_type(16))) _Float16 v16h;
typedef __attribute__((ext_vector_type(8)))  float    v8f;

#define B_  2
#define T_  2048
#define D_  1024
#define H_  16
#define HD_ 64

union Frag16 { v16h v; uint32_t u[8]; };

// A-matrix 16x32 f16 fragment from row-major [*, ld] at p (element (m,k) = p[m*ld+k]).
// ISA 7.12.2: lanes 0-15 -> M=0..15; VGPR v holds K = (v>>2)*16 + half*8 + (v&3)*2, +1
__device__ __forceinline__ v16h load_a16(const f16* p, int ld) {
  int lane = threadIdx.x & 31;
  int m = lane & 15, hl = lane >> 4;
  const uint32_t* row = (const uint32_t*)(p + (size_t)m * ld);
  Frag16 f;
#pragma unroll
  for (int v = 0; v < 8; ++v) {
    int kpair = (((v >> 2) * 16 + hl * 8 + (v & 3) * 2) >> 1);
    f.u[v] = row[kpair];
  }
  return f.v;
}

// B-matrix 32x16 f16 fragment, sourced from B^T stored row-major [N, ld] (element (k,n) = pT[n*ld+k]).
// Lane holds column n = lane&15; VGPR v holds K = (v>>2)*16 + half*8 + (v&3)*2, +1
__device__ __forceinline__ v16h load_bT16(const f16* pT, int ld) {
  int lane = threadIdx.x & 31;
  int n = lane & 15, hl = lane >> 4;
  const uint32_t* row = (const uint32_t*)(pT + (size_t)n * ld);
  Frag16 f;
#pragma unroll
  for (int v = 0; v < 8; ++v) {
    f.u[v] = row[(v >> 2) * 8 + hl * 4 + (v & 3)];
  }
  return f.v;
}

__device__ __forceinline__ float half16_max(float x) {
  x = fmaxf(x, __shfl_xor(x, 1));
  x = fmaxf(x, __shfl_xor(x, 2));
  x = fmaxf(x, __shfl_xor(x, 4));
  x = fmaxf(x, __shfl_xor(x, 8));
  return x;
}
__device__ __forceinline__ float half16_sum(float x) {
  x += __shfl_xor(x, 1);
  x += __shfl_xor(x, 2);
  x += __shfl_xor(x, 4);
  x += __shfl_xor(x, 8);
  return x;
}

// ---------------- conversion kernels ----------------
__global__ void cvt_f32_f16_v4(const float* __restrict__ src, f16* __restrict__ dst, int n4) {
  int i = blockIdx.x * blockDim.x + threadIdx.x;
  if (i < n4) {
    const float4 v = ((const float4*)src)[i];
    union { f16 h[4]; uint2 u; } o;
    o.h[0] = (f16)v.x; o.h[1] = (f16)v.y; o.h[2] = (f16)v.z; o.h[3] = (f16)v.w;
    ((uint2*)dst)[i] = o.u;
  }
}

// src [K,N] f32 -> dst [N,K] f16 via 32x32 LDS tile (both sides coalesced)
__global__ __launch_bounds__(256) void cvt_transpose_tiled(const float* __restrict__ src,
                                                           f16* __restrict__ dst, int K, int N) {
  __shared__ f16 tile[32][33];
  int tx = threadIdx.x & 31, ty = threadIdx.x >> 5;   // 32 x 8
  int k0 = blockIdx.x * 32;                           // along K (rows of src)
  int n0 = blockIdx.y * 32;                           // along N (cols of src)
#pragma unroll
  for (int i = ty; i < 32; i += 8)
    tile[i][tx] = (f16)src[(size_t)(k0 + i) * N + n0 + tx];
  __syncthreads();
#pragma unroll
  for (int i = ty; i < 32; i += 8)
    dst[(size_t)(n0 + i) * K + k0 + tx] = tile[tx][i];
}

// ---------------- QKV projection: 32x64 tile per wave ----------------
__global__ __launch_bounds__(256) void qkv_gemm(const f16* __restrict__ xh,
                                                const f16* __restrict__ wT,
                                                const float* __restrict__ bias,
                                                f16* __restrict__ Qh, f16* __restrict__ Kh,
                                                f16* __restrict__ Vth) {
  int lane = threadIdx.x & 31;
  int wave = threadIdx.x >> 5;
  int tile = blockIdx.x * 8 + wave;
  const int NT = (3 * D_) / 64;  // 48 n-tiles of width 64
  int m0 = (tile / NT) * 32;
  int n0 = (tile % NT) * 64;
  const f16* A0 = xh + (size_t)m0 * D_;
  const f16* A1 = xh + (size_t)(m0 + 16) * D_;
  v8f acc[2][4] = {};
#pragma unroll 2
  for (int k = 0; k < D_; k += 32) {
    v16h a0 = load_a16(A0 + k, D_);
    v16h a1 = load_a16(A1 + k, D_);
#pragma unroll
    for (int j = 0; j < 4; ++j) {
      v16h b = load_bT16(wT + (size_t)(n0 + 16 * j) * D_ + k, D_);
      acc[0][j] = __builtin_amdgcn_wmma_f32_16x16x32_f16(false, a0, false, b, (short)0, acc[0][j], false, false);
      acc[1][j] = __builtin_amdgcn_wmma_f32_16x16x32_f16(false, a1, false, b, (short)0, acc[1][j], false, false);
    }
  }
  int hl = lane >> 4, nn = lane & 15;
#pragma unroll
  for (int j = 0; j < 4; ++j) {
    int c = n0 + 16 * j + nn;
    float bv = bias[c];
    int s = c >> 10;
    int rem = c & (D_ - 1);
    int h = rem >> 6, e = rem & (HD_ - 1);
#pragma unroll
    for (int i = 0; i < 2; ++i) {
#pragma unroll
      for (int r = 0; r < 8; ++r) {
        int tg = m0 + 16 * i + r + 8 * hl;     // global token index in [0, B*T)
        int bb = tg >> 11, t = tg & (T_ - 1);
        f16 val = (f16)(acc[i][j][r] + bv);
        size_t bh = (size_t)bb * H_ + h;
        if (s == 0)      Qh [(bh * T_ + t) * HD_ + e] = val;
        else if (s == 1) Kh [(bh * T_ + t) * HD_ + e] = val;
        else             Vth[(bh * HD_ + e) * T_ + t] = val;
      }
    }
  }
}

// ---------------- causal flash attention: 32 queries x HD=64 per wave ----------------
__global__ __launch_bounds__(128) void flash_attn(const f16* __restrict__ Qh,
                                                  const f16* __restrict__ Kh,
                                                  const f16* __restrict__ Vth,
                                                  f16* __restrict__ attnh) {
  __shared__ __align__(16) f16 pbuf[4][2][16 * 32];
  int lane = threadIdx.x & 31;
  int w = threadIdx.x >> 5;
  // grid: B*H*(T/128); block = 4 waves = 128 query rows
  int qt = blockIdx.x & 15;            // T/128 = 16
  int bh = blockIdx.x >> 4;
  int b = bh >> 4, h = bh & 15;
  int q0 = qt * 128 + w * 32;          // multiple of 32 -> kt <= q0 for all tiles

  const f16* Qb = Qh + ((size_t)(b * H_ + h) * T_ + q0) * HD_;
  const f16* Kb = Kh + (size_t)(b * H_ + h) * T_ * HD_;
  const f16* Vb = Vth + (size_t)(b * H_ + h) * HD_ * T_;

  v16h aq[2][2];
  aq[0][0] = load_a16(Qb, HD_);                 // rows q0..q0+15, hd 0..31
  aq[0][1] = load_a16(Qb + 32, HD_);            // hd 32..63
  aq[1][0] = load_a16(Qb + 16 * HD_, HD_);      // rows q0+16..q0+31
  aq[1][1] = load_a16(Qb + 16 * HD_ + 32, HD_);

  int hl = lane >> 4, nn = lane & 15;
  float mrow[2][8], lrow[2][8];
  v8f o[2][4] = {};
#pragma unroll
  for (int i = 0; i < 2; ++i)
#pragma unroll
    for (int r = 0; r < 8; ++r) { mrow[i][r] = -1e30f; lrow[i][r] = 0.0f; }

  for (int kt = 0; kt < q0 + 32; kt += 32) {
    if (kt + 32 <= q0) {
      // one lane per future key row: 32 lanes cover the whole next 32x128B K tile
      __builtin_prefetch(Kb + (size_t)(kt + 32 + lane) * HD_, 0, 3);
    }
    // ---- K fragments for 32 keys (shared across both query row-fragments) ----
    v16h bk00 = load_bT16(Kb + (size_t)kt * HD_, HD_);
    v16h bk01 = load_bT16(Kb + (size_t)kt * HD_ + 32, HD_);
    v16h bk10 = load_bT16(Kb + (size_t)(kt + 16) * HD_, HD_);
    v16h bk11 = load_bT16(Kb + (size_t)(kt + 16) * HD_ + 32, HD_);
    v8f s[2][2] = {};
#pragma unroll
    for (int i = 0; i < 2; ++i) {
      s[i][0] = __builtin_amdgcn_wmma_f32_16x16x32_f16(false, aq[i][0], false, bk00, (short)0, s[i][0], false, false);
      s[i][0] = __builtin_amdgcn_wmma_f32_16x16x32_f16(false, aq[i][1], false, bk01, (short)0, s[i][0], false, false);
      s[i][1] = __builtin_amdgcn_wmma_f32_16x16x32_f16(false, aq[i][0], false, bk10, (short)0, s[i][1], false, false);
      s[i][1] = __builtin_amdgcn_wmma_f32_16x16x32_f16(false, aq[i][1], false, bk11, (short)0, s[i][1], false, false);
    }
    // ---- online softmax (C layout: VGPR r <-> row r + 8*half, col = lane&15) ----
#pragma unroll
    for (int i = 0; i < 2; ++i) {
#pragma unroll
      for (int r = 0; r < 8; ++r) {
        int q = q0 + 16 * i + r + 8 * hl;
        float slo = s[i][0][r] * 0.125f;   // 1/sqrt(64)
        float shi = s[i][1][r] * 0.125f;
        if (kt + nn > q)      slo = -1e30f;
        if (kt + 16 + nn > q) shi = -1e30f;
        float tmax = half16_max(fmaxf(slo, shi));
        float mnew = fmaxf(mrow[i][r], tmax);
        float alpha = __expf(mrow[i][r] - mnew);
        float plo = __expf(slo - mnew);
        float phi = __expf(shi - mnew);
        lrow[i][r] = lrow[i][r] * alpha + half16_sum(plo + phi);
        mrow[i][r] = mnew;
#pragma unroll
        for (int j = 0; j < 4; ++j) o[i][j][r] *= alpha;
        int row = r + 8 * hl;
        pbuf[w][i][row * 32 + nn]      = (f16)plo;
        pbuf[w][i][row * 32 + 16 + nn] = (f16)phi;
      }
    }
    // re-load P in A-fragment layout (DS ops from the same wave are in-order)
    v16h ap0 = load_a16(pbuf[w][0], 32);
    v16h ap1 = load_a16(pbuf[w][1], 32);
    // ---- O += P V (V^T buffer: element (hd, t)); V frags shared across row-frags ----
#pragma unroll
    for (int j = 0; j < 4; ++j) {
      v16h bv = load_bT16(Vb + (size_t)j * 16 * T_ + kt, T_);
      o[0][j] = __builtin_amdgcn_wmma_f32_16x16x32_f16(false, ap0, false, bv, (short)0, o[0][j], false, false);
      o[1][j] = __builtin_amdgcn_wmma_f32_16x16x32_f16(false, ap1, false, bv, (short)0, o[1][j], false, false);
    }
  }

  // ---- finalize: O / l, store f16 attn output [B,T,D] ----
#pragma unroll
  for (int i = 0; i < 2; ++i) {
#pragma unroll
    for (int r = 0; r < 8; ++r) {
      int q = q0 + 16 * i + r + 8 * hl;
      float inv = 1.0f / lrow[i][r];
      size_t base = ((size_t)b * T_ + q) * D_ + (size_t)h * HD_ + nn;
#pragma unroll
      for (int j = 0; j < 4; ++j)
        attnh[base + 16 * j] = (f16)(o[i][j][r] * inv);
    }
  }
}

// ---------------- output projection: 32x64 tile per wave ----------------
__global__ __launch_bounds__(256) void out_gemm(const f16* __restrict__ attnh,
                                                const f16* __restrict__ wT,
                                                const float* __restrict__ bias,
                                                float* __restrict__ out) {
  int lane = threadIdx.x & 31;
  int wave = threadIdx.x >> 5;
  int tile = blockIdx.x * 8 + wave;
  const int NT = D_ / 64;  // 16
  int m0 = (tile / NT) * 32;
  int n0 = (tile % NT) * 64;
  const f16* A0 = attnh + (size_t)m0 * D_;
  const f16* A1 = attnh + (size_t)(m0 + 16) * D_;
  v8f acc[2][4] = {};
#pragma unroll 2
  for (int k = 0; k < D_; k += 32) {
    v16h a0 = load_a16(A0 + k, D_);
    v16h a1 = load_a16(A1 + k, D_);
#pragma unroll
    for (int j = 0; j < 4; ++j) {
      v16h b = load_bT16(wT + (size_t)(n0 + 16 * j) * D_ + k, D_);
      acc[0][j] = __builtin_amdgcn_wmma_f32_16x16x32_f16(false, a0, false, b, (short)0, acc[0][j], false, false);
      acc[1][j] = __builtin_amdgcn_wmma_f32_16x16x32_f16(false, a1, false, b, (short)0, acc[1][j], false, false);
    }
  }
  int hl = lane >> 4, nn = lane & 15;
#pragma unroll
  for (int j = 0; j < 4; ++j) {
    float bv = bias[n0 + 16 * j + nn];
#pragma unroll
    for (int i = 0; i < 2; ++i) {
#pragma unroll
      for (int r = 0; r < 8; ++r) {
        int tg = m0 + 16 * i + r + 8 * hl;
        out[(size_t)tg * D_ + n0 + 16 * j + nn] = acc[i][j][r] + bv;
      }
    }
  }
}

extern "C" void kernel_launch(void* const* d_in, const int* in_sizes, int n_in,
                              void* d_out, int out_size, void* d_ws, size_t ws_size,
                              hipStream_t stream) {
  (void)in_sizes; (void)n_in; (void)out_size; (void)ws_size;
  const float* x    = (const float*)d_in[0];
  const float* Wqkv = (const float*)d_in[1];
  const float* bqkv = (const float*)d_in[2];
  const float* Wout = (const float*)d_in[3];
  const float* bout = (const float*)d_in[4];
  float* out = (float*)d_out;

  char* ws = (char*)d_ws;
  size_t off = 0;
  f16* xh     = (f16*)(ws + off); off += (size_t)B_ * T_ * D_ * sizeof(f16);        // 8 MiB
  f16* WqkvT  = (f16*)(ws + off); off += (size_t)3 * D_ * D_ * sizeof(f16);         // 6 MiB
  f16* WoutT  = (f16*)(ws + off); off += (size_t)D_ * D_ * sizeof(f16);             // 2 MiB
  f16* Qh     = (f16*)(ws + off); off += (size_t)B_ * H_ * T_ * HD_ * sizeof(f16);  // 8 MiB
  f16* Kh     = (f16*)(ws + off); off += (size_t)B_ * H_ * T_ * HD_ * sizeof(f16);  // 8 MiB
  f16* Vth    = (f16*)(ws + off); off += (size_t)B_ * H_ * T_ * HD_ * sizeof(f16);  // 8 MiB
  f16* attnh  = (f16*)(ws + off); off += (size_t)B_ * T_ * D_ * sizeof(f16);        // 8 MiB

  const int nx4 = (B_ * T_ * D_) / 4;  // 1048576 float4 groups

  cvt_f32_f16_v4<<<(nx4 + 255) / 256, 256, 0, stream>>>(x, xh, nx4);
  cvt_transpose_tiled<<<dim3(D_ / 32, (3 * D_) / 32), 256, 0, stream>>>(Wqkv, WqkvT, D_, 3 * D_);
  cvt_transpose_tiled<<<dim3(D_ / 32, D_ / 32), 256, 0, stream>>>(Wout, WoutT, D_, D_);

  // (B*T/32) * (3D/64) tiles / 8 waves per block = 768 blocks
  qkv_gemm<<<(128 * 48) / 8, 256, 0, stream>>>(xh, WqkvT, bqkv, Qh, Kh, Vth);

  // B*H*(T/128) = 512 blocks, 4 waves each
  flash_attn<<<B_ * H_ * (T_ / 128), 128, 0, stream>>>(Qh, Kh, Vth, attnh);

  // (B*T/32) * (D/64) tiles / 8 = 256 blocks
  out_gemm<<<(128 * 16) / 8, 256, 0, stream>>>(attnh, WoutT, bout, out);
}
